// GRecBillshock_90426241450661
// MI455X (gfx1250) — matrix operand order repaired and checked by reading, not compiled, not run on hardware
//
#include <hip/hip_runtime.h>
#include <hip/hip_bf16.h>
#include <math.h>

// ---------------- types ----------------
typedef __bf16 bf16;
typedef __attribute__((ext_vector_type(16))) __bf16 v16bf;
typedef __attribute__((ext_vector_type(8)))  __bf16 bf16x8;
typedef __attribute__((ext_vector_type(4)))  __bf16 bf16x4;
typedef __attribute__((ext_vector_type(8)))  float  v8f;
typedef __attribute__((ext_vector_type(4)))  float  f32x4;

// ---------------- problem constants ----------------
#define BQ   16384      // batch (tokens)
#define DD   1024       // model dim
#define NEXP 16         // experts
#define CAP  4096       // capacity
#define NWID 64         // wide features
#define DHALF 512       // D/2
#define N3   3072       // 3*D
#define VV   100000     // vocab
#define EE   128        // embed dim
#define FF   8          // fields

// GEMM tiling: block tile 128(M) x 256(N), 8 waves (2x4), wave tile 64x64
#define BM 128
#define BNT 256
#define BK 64
#define LDT 72          // padded LDS K-stride (elements), 144B rows (16B aligned)
#define STAGE_ELEMS ((BM + BNT) * LDT)
#define GEMM_SMEM (2 * STAGE_ELEMS * (int)sizeof(bf16))   // 110592 B

__device__ __forceinline__ float leakyf(float x){ return x >= 0.f ? x : 0.2f * x; }

// =====================================================================
// Generic batched GEMM: C[M,N] = act(A[M,K] @ Bt[N,K]^T + bias), bf16 in,
// f32 accumulate via v_wmma_f32_16x16x32_bf16, bf16 or f32 out.
// Grid: (N/BNT, M/BM, batch). Block: 256.
// Per 32-K step, per wave: 16 ds_load_b128 -> 16 v_wmma (1:1).
// =====================================================================
__global__ __launch_bounds__(256)
void gemm_bf16_wmma(const bf16* __restrict__ Aall, const bf16* __restrict__ Btall,
                    const float* __restrict__ biasAll,
                    bf16* __restrict__ Cb, float* __restrict__ Cf,
                    int M, int N, int K, int ldc,
                    long long sA, long long sB, long long sBias, long long sC,
                    int doLeaky)
{
    extern __shared__ char smem[];
    bf16* lds = (bf16*)smem;

    const int tid  = threadIdx.x;
    const int bn   = blockIdx.x;
    const int bm   = blockIdx.y;
    const int bz   = blockIdx.z;

    const bf16* A  = Aall  + (long long)bz * sA + (long long)bm * BM * K;
    const bf16* Bt = Btall + (long long)bz * sB + (long long)bn * BNT * K;
    const float* bias = biasAll ? (biasAll + (long long)bz * sBias) : nullptr;

    const int lane = tid & 31;
    const int wave = tid >> 5;
    const int wm   = wave >> 2;   // 0..1
    const int wn   = wave & 3;    // 0..3
    const int half = lane >> 4;   // 0/1
    const int l16  = lane & 15;

    v8f acc[4][4];
    const v8f zero8 = {0.f,0.f,0.f,0.f,0.f,0.f,0.f,0.f};
    #pragma unroll
    for (int i = 0; i < 4; ++i)
        #pragma unroll
        for (int j = 0; j < 4; ++j) acc[i][j] = zero8;

    const int nStages = K / BK;

    auto loadStage = [&](int s, int buf) {
        bf16* la = lds + buf * STAGE_ELEMS;
        bf16* lb = la + BM * LDT;
        const int ks = s * BK;
        #pragma unroll
        for (int i = 0; i < 4; ++i) {                 // A: 128x64
            int c   = tid * 4 + i;                    // 0..1023
            int lin = c * 8;
            int row = lin >> 6;
            int col = lin & 63;
            bf16x8 av = *(const bf16x8*)(A + (long long)row * K + ks + col);
            *(bf16x8*)(la + row * LDT + col) = av;
        }
        #pragma unroll
        for (int i = 0; i < 8; ++i) {                 // B: 256x64
            int c   = tid * 8 + i;                    // 0..2047
            int lin = c * 8;
            int row = lin >> 6;
            int col = lin & 63;
            bf16x8 bv = *(const bf16x8*)(Bt + (long long)row * K + ks + col);
            *(bf16x8*)(lb + row * LDT + col) = bv;
        }
        if (s + 1 < nStages) {
            __builtin_prefetch(A  + (long long)(tid >> 1) * K + ks + BK, 0, 1);
            __builtin_prefetch(Bt + (long long)tid * K + ks + BK, 0, 1);
        }
    };

    loadStage(0, 0);
    __syncthreads();

    for (int s = 0; s < nStages; ++s) {
        const int buf = s & 1;
        if (s + 1 < nStages) loadStage(s + 1, buf ^ 1);

        bf16* la = lds + buf * STAGE_ELEMS;
        bf16* lb = la + BM * LDT;

        #pragma unroll
        for (int kk = 0; kk < BK; kk += 32) {
            v16bf af[4], bfm[4];
            #pragma unroll
            for (int mt = 0; mt < 4; ++mt) {
                int row = wm * 64 + mt * 16 + l16;
                const bf16* p = la + row * LDT + kk + half * 8;
                union { v16bf v; bf16x8 h[2]; } u;
                u.h[0] = *(const bf16x8*)p;         // K = kk + half*8 + 0..7
                u.h[1] = *(const bf16x8*)(p + 16);  // K = kk + 16 + half*8 + 0..7
                af[mt] = u.v;
            }
            #pragma unroll
            for (int nt = 0; nt < 4; ++nt) {
                int rowb = wn * 64 + nt * 16 + l16;
                const bf16* p = lb + rowb * LDT + kk + half * 8;
                union { v16bf v; bf16x8 h[2]; } u;
                u.h[0] = *(const bf16x8*)p;         // K = kk + half*8 + 0..7
                u.h[1] = *(const bf16x8*)(p + 16);  // K = kk + 16 + half*8 + 0..7
                bfm[nt] = u.v;
            }
            #pragma unroll
            for (int mt = 0; mt < 4; ++mt)
                #pragma unroll
                for (int nt = 0; nt < 4; ++nt)
                    acc[mt][nt] = __builtin_amdgcn_wmma_f32_16x16x32_bf16(
                        false, af[mt], false, bfm[nt], (short)0, acc[mt][nt], false, false);
        }
        __syncthreads();
    }

    const long long cbase = (long long)bz * sC;
    #pragma unroll
    for (int mt = 0; mt < 4; ++mt) {
        #pragma unroll
        for (int nt = 0; nt < 4; ++nt) {
            int col = bn * BNT + wn * 64 + nt * 16 + l16;
            float bv = bias ? bias[col] : 0.f;
            #pragma unroll
            for (int v = 0; v < 8; ++v) {
                int row = bm * BM + wm * 64 + mt * 16 + half * 8 + v;
                float val = acc[mt][nt][v] + bv;
                if (doLeaky) val = leakyf(val);
                long long off = cbase + (long long)row * ldc + col;
                if (Cf) Cf[off] = val;
                else    Cb[off] = (bf16)val;
            }
        }
    }
}

// =====================================================================
// Fused embedding gather + deep GEMM: ctx[:, 0:512] =
//   leaky(concat_emb(16384x1024) @ deepWt^T + deep_b), bf16 out (ldc=1024)
// Gathers f32 embedding rows straight into LDS A-tile as bf16.
// Grid: (512/BNT, 16384/BM). Block 256.
// =====================================================================
__global__ __launch_bounds__(256)
void embed_gemm_wmma(const int* __restrict__ deep_in, const float* __restrict__ embed,
                     const bf16* __restrict__ Btall, const float* __restrict__ bias,
                     bf16* __restrict__ Cb)
{
    extern __shared__ char smem[];
    bf16* lds  = (bf16*)smem;
    int*  lidx = (int*)(smem + GEMM_SMEM);   // 128 tokens x 8 fields

    const int tid = threadIdx.x;
    const int bn  = blockIdx.x;
    const int bm  = blockIdx.y;
    const int K   = DD;

    const bf16* Bt = Btall + (long long)bn * BNT * K;

    #pragma unroll
    for (int i = 0; i < 4; ++i) {
        int j = tid * 4 + i;                 // 0..1023
        lidx[j] = deep_in[(size_t)bm * BM * FF + j];
    }
    __syncthreads();

    const int lane = tid & 31;
    const int wave = tid >> 5;
    const int wm   = wave >> 2;
    const int wn   = wave & 3;
    const int half = lane >> 4;
    const int l16  = lane & 15;

    v8f acc[4][4];
    const v8f zero8 = {0.f,0.f,0.f,0.f,0.f,0.f,0.f,0.f};
    #pragma unroll
    for (int i = 0; i < 4; ++i)
        #pragma unroll
        for (int j = 0; j < 4; ++j) acc[i][j] = zero8;

    const int nStages = K / BK;   // 16

    auto loadStage = [&](int s, int buf) {
        bf16* la = lds + buf * STAGE_ELEMS;
        bf16* lb = la + BM * LDT;
        const int ks = s * BK;
        const int f       = ks >> 7;     // field, uniform per stage
        const int offBase = ks & 127;    // 0 or 64
        #pragma unroll
        for (int i = 0; i < 4; ++i) {                 // A via gather: 128x64
            int c   = tid * 4 + i;
            int lin = c * 8;
            int row = lin >> 6;
            int col = lin & 63;
            int idx = lidx[row * FF + f];
            const float* src = embed + ((size_t)f * VV + (size_t)idx) * EE + offBase + col;
            f32x4 x0 = *(const f32x4*)src;
            f32x4 x1 = *(const f32x4*)(src + 4);
            bf16x8 av;
            av[0]=(bf16)x0[0]; av[1]=(bf16)x0[1]; av[2]=(bf16)x0[2]; av[3]=(bf16)x0[3];
            av[4]=(bf16)x1[0]; av[5]=(bf16)x1[1]; av[6]=(bf16)x1[2]; av[7]=(bf16)x1[3];
            *(bf16x8*)(la + row * LDT + col) = av;
        }
        #pragma unroll
        for (int i = 0; i < 8; ++i) {                 // B: 256x64
            int c   = tid * 8 + i;
            int lin = c * 8;
            int row = lin >> 6;
            int col = lin & 63;
            bf16x8 bv = *(const bf16x8*)(Bt + (long long)row * K + ks + col);
            *(bf16x8*)(lb + row * LDT + col) = bv;
        }
    };

    loadStage(0, 0);
    __syncthreads();

    for (int s = 0; s < nStages; ++s) {
        const int buf = s & 1;
        if (s + 1 < nStages) loadStage(s + 1, buf ^ 1);
        bf16* la = lds + buf * STAGE_ELEMS;
        bf16* lb = la + BM * LDT;
        #pragma unroll
        for (int kk = 0; kk < BK; kk += 32) {
            v16bf af[4], bfm[4];
            #pragma unroll
            for (int mt = 0; mt < 4; ++mt) {
                int row = wm * 64 + mt * 16 + l16;
                const bf16* p = la + row * LDT + kk + half * 8;
                union { v16bf v; bf16x8 h[2]; } u;
                u.h[0] = *(const bf16x8*)p;
                u.h[1] = *(const bf16x8*)(p + 16);
                af[mt] = u.v;
            }
            #pragma unroll
            for (int nt = 0; nt < 4; ++nt) {
                int rowb = wn * 64 + nt * 16 + l16;
                const bf16* p = lb + rowb * LDT + kk + half * 8;
                union { v16bf v; bf16x8 h[2]; } u;
                u.h[0] = *(const bf16x8*)p;
                u.h[1] = *(const bf16x8*)(p + 16);
                bfm[nt] = u.v;
            }
            #pragma unroll
            for (int mt = 0; mt < 4; ++mt)
                #pragma unroll
                for (int nt = 0; nt < 4; ++nt)
                    acc[mt][nt] = __builtin_amdgcn_wmma_f32_16x16x32_bf16(
                        false, af[mt], false, bfm[nt], (short)0, acc[mt][nt], false, false);
        }
        __syncthreads();
    }

    #pragma unroll
    for (int mt = 0; mt < 4; ++mt) {
        #pragma unroll
        for (int nt = 0; nt < 4; ++nt) {
            int col = bn * BNT + wn * 64 + nt * 16 + l16;   // 0..511
            float bv = bias[col];
            #pragma unroll
            for (int v = 0; v < 8; ++v) {
                int row = bm * BM + wm * 64 + mt * 16 + half * 8 + v;
                float val = leakyf(acc[mt][nt][v] + bv);
                Cb[(long long)row * DD + col] = (bf16)val;  // deep half of ctx
            }
        }
    }
}

// =====================================================================
// Weight transpose + f32->bf16:  dst[N,K] = (bf16) src[K,N]^T  (batched via z)
// =====================================================================
__global__ void transpose_f32_to_bf16(const float* __restrict__ src, bf16* __restrict__ dst,
                                      int K, int N, long long sSrc, long long sDst)
{
    __shared__ float tile[32][33];
    const float* S = src + (long long)blockIdx.z * sSrc;
    bf16* Dp = dst + (long long)blockIdx.z * sDst;
    int tx = threadIdx.x, ty = threadIdx.y;
    int n0 = blockIdx.x * 32, k0 = blockIdx.y * 32;
    #pragma unroll
    for (int i = 0; i < 32; i += 8)
        tile[ty + i][tx] = S[(long long)(k0 + ty + i) * N + n0 + tx];
    __syncthreads();
    #pragma unroll
    for (int i = 0; i < 32; i += 8)
        Dp[(long long)(n0 + ty + i) * K + k0 + tx] = (bf16)tile[tx][ty + i];
}

// =====================================================================
// Wide branch: ctx[:, 512:1024] = leaky(BN(wide_in @ wide_W + b)). Block=128.
// =====================================================================
__global__ void wide_kernel(const float* __restrict__ wide_in, const bf16* __restrict__ wideWt,
                            const float* __restrict__ wide_b, const float* __restrict__ gamma,
                            const float* __restrict__ beta, bf16* __restrict__ ctx)
{
    int t = blockIdx.x;
    __shared__ float w[NWID];
    int tid = threadIdx.x;
    if (tid < NWID) w[tid] = wide_in[(size_t)t * NWID + tid];
    __syncthreads();
    const float inv = 0.999995000037f;  // 1/sqrt(1+1e-5)
    #pragma unroll
    for (int i = 0; i < 4; ++i) {
        int n = tid * 4 + i;            // 0..511
        const bf16* wr = wideWt + (size_t)n * NWID;
        float acc = 0.f;
        #pragma unroll 8
        for (int k = 0; k < NWID; ++k) acc += w[k] * (float)wr[k];
        float val = acc + wide_b[n];
        val = gamma[n] * val * inv + beta[n];
        ctx[(size_t)t * DD + DHALF + n] = (bf16)leakyf(val);
    }
}

// =====================================================================
// Router: softmax(ctx @ router_W) -> probs[t,16], top-2 (k-major flat layout).
// Block=128 (16 experts x 8 K-slices).
// =====================================================================
__global__ void router_kernel(const bf16* __restrict__ ctx, const float* __restrict__ rW,
                              float* __restrict__ probs, float* __restrict__ topv,
                              int* __restrict__ topi)
{
    int t = blockIdx.x;
    int tid = threadIdx.x;
    int e = tid & 15, sl = tid >> 4;
    const bf16* x = ctx + (size_t)t * DD;
    float acc = 0.f;
    for (int k = sl * 128; k < sl * 128 + 128; ++k)
        acc += (float)x[k] * rW[(size_t)k * NEXP + e];
    __shared__ float part[128];
    __shared__ float logits[NEXP];
    part[tid] = acc;
    __syncthreads();
    if (tid < NEXP) {
        float s = 0.f;
        #pragma unroll
        for (int j = 0; j < 8; ++j) s += part[j * NEXP + tid];
        logits[tid] = s;
    }
    __syncthreads();
    if (tid == 0) {
        float mx = logits[0];
        for (int i = 1; i < NEXP; ++i) mx = fmaxf(mx, logits[i]);
        float ex[NEXP]; float sum = 0.f;
        for (int i = 0; i < NEXP; ++i) { ex[i] = expf(logits[i] - mx); sum += ex[i]; }
        float is = 1.f / sum;
        int i0 = 0; float v0 = -1.f;
        for (int i = 0; i < NEXP; ++i) {
            float p = ex[i] * is;
            probs[(size_t)t * NEXP + i] = p;
            if (p > v0) { v0 = p; i0 = i; }
        }
        int i1 = 0; float v1 = -1.f;
        for (int i = 0; i < NEXP; ++i) {
            if (i == i0) continue;
            float p = ex[i] * is;
            if (p > v1) { v1 = p; i1 = i; }
        }
        topv[t] = v0;  topv[BQ + t] = v1;
        topi[t] = i0;  topi[BQ + t] = i1;
    }
}

// =====================================================================
// Single-block capacity scan over 32768 assignments (k-major order),
// wave32 ballots for intra-chunk ranks. slot[j] = e*CAP+pos or -1.
// =====================================================================
__global__ void scan_kernel(const int* __restrict__ flatE, int* __restrict__ slot,
                            int* __restrict__ counts)
{
    __shared__ int wcnt[32][NEXP];
    __shared__ int chunkTot[NEXP];
    __shared__ int base[NEXP];
    int tid = threadIdx.x, lane = tid & 31, w = tid >> 5;
    if (tid < NEXP) base[tid] = 0;
    __syncthreads();
    unsigned lmask = (1u << lane) - 1u;
    for (int c = 0; c < (2 * BQ) / 1024; ++c) {
        int j = c * 1024 + tid;
        int e = flatE[j];
        int lanerank = 0;
        #pragma unroll
        for (int ex = 0; ex < NEXP; ++ex) {
            unsigned m = __builtin_amdgcn_ballot_w32(e == ex);
            if (ex == e) lanerank = __popc(m & lmask);
            if (lane == 0) wcnt[w][ex] = __popc(m);
        }
        __syncthreads();
        if (tid < NEXP) {
            int run = 0;
            for (int ww = 0; ww < 32; ++ww) { int v = wcnt[ww][tid]; wcnt[ww][tid] = run; run += v; }
            chunkTot[tid] = run;
        }
        __syncthreads();
        int pos = base[e] + wcnt[w][e] + lanerank;
        slot[j] = (pos < CAP) ? (e * CAP + pos) : -1;
        __syncthreads();
        if (tid < NEXP) base[tid] += chunkTot[tid];
        __syncthreads();
    }
    if (tid < NEXP) counts[tid] = base[tid];
}

// =====================================================================
// Utility: zero bf16 region (8 elements / thread).
// =====================================================================
__global__ void zero_bf16(bf16* __restrict__ p, long long n8)
{
    long long i = (long long)blockIdx.x * blockDim.x + threadIdx.x;
    if (i < n8) { bf16x8 z = {}; *(bf16x8*)(p + i * 8) = z; }
}

// =====================================================================
// Scatter ctx rows into expert capacity buffer. Grid = 32768, block 256.
// =====================================================================
__global__ void scatter_kernel(const bf16* __restrict__ ctx, const int* __restrict__ slot,
                               bf16* __restrict__ buf)
{
    int j = blockIdx.x;
    int s = slot[j];
    if (s < 0) return;
    int t = j & (BQ - 1);
    const bf16x4* src = (const bf16x4*)(ctx + (size_t)t * DD);
    bf16x4* dst = (bf16x4*)(buf + (size_t)s * DD);
    dst[threadIdx.x] = src[threadIdx.x];
}

// =====================================================================
// Combine: moe[t] = w0*y[slot0] + w1*y[slot1] (bf16 out). Grid 16384, blk 256.
// =====================================================================
__global__ void combine_kernel(const bf16* __restrict__ y, const int* __restrict__ slot,
                               const float* __restrict__ topv, bf16* __restrict__ moe)
{
    int t = blockIdx.x, tid = threadIdx.x;
    int s0 = slot[t], s1 = slot[BQ + t];
    float w0 = topv[t], w1 = topv[BQ + t];
    #pragma unroll
    for (int i = 0; i < 4; ++i) {
        int d = tid * 4 + i;
        float a = 0.f;
        if (s0 >= 0) a += w0 * (float)y[(size_t)s0 * DD + d];
        if (s1 >= 0) a += w1 * (float)y[(size_t)s1 * DD + d];
        moe[(size_t)t * DD + d] = (bf16)a;
    }
}

// =====================================================================
// Final head: out[t] = sigmoid(h2[t,:] . out_W + out_b). Grid 16384, blk 256.
// =====================================================================
__global__ void final_kernel(const float* __restrict__ h2, const float* __restrict__ outW,
                             const float* __restrict__ outb, float* __restrict__ out)
{
    int t = blockIdx.x, tid = threadIdx.x;
    float acc = 0.f;
    for (int d = tid; d < DD; d += 256) acc += h2[(size_t)t * DD + d] * outW[d];
    __shared__ float red[256];
    red[tid] = acc; __syncthreads();
    for (int s = 128; s > 0; s >>= 1) { if (tid < s) red[tid] += red[tid + s]; __syncthreads(); }
    if (tid == 0) {
        float z = red[0] + outb[0];
        out[t] = 1.f / (1.f + expf(-z));
    }
}

// =====================================================================
// Aux loss: 16 * sum_e (counts_e / 32768) * mean_t probs[t,e]. 1 block, 256.
// =====================================================================
__global__ void aux_kernel(const float* __restrict__ probs, const int* __restrict__ counts,
                           float* __restrict__ auxOut)
{
    int tid = threadIdx.x;
    int e = tid & 15, p = tid >> 4;
    float s = 0.f;
    for (int t = p; t < BQ; t += 16) s += probs[(size_t)t * NEXP + e];
    __shared__ float red[256];
    red[tid] = s; __syncthreads();
    if (tid < NEXP) {
        float tot = 0.f;
        for (int pp = 0; pp < 16; ++pp) tot += red[pp * 16 + tid];
        red[tid] = tot;
    }
    __syncthreads();
    if (tid == 0) {
        float aux = 0.f;
        for (int ee = 0; ee < NEXP; ++ee) {
            float meanp = red[ee] / (float)BQ;
            float frac  = (float)counts[ee] / (float)(2 * BQ);
            aux += frac * meanp;
        }
        auxOut[0] = (float)NEXP * aux;
    }
}

// =====================================================================
// Host launcher
// =====================================================================
extern "C" void kernel_launch(void* const* d_in, const int* in_sizes, int n_in,
                              void* d_out, int out_size, void* d_ws, size_t ws_size,
                              hipStream_t stream)
{
    (void)in_sizes; (void)n_in; (void)out_size; (void)ws_size;

    const int*   deep_in  = (const int*)  d_in[0];
    const float* wide_in  = (const float*)d_in[1];
    const float* embed    = (const float*)d_in[2];
    const float* deep_W   = (const float*)d_in[3];
    const float* deep_b   = (const float*)d_in[4];
    const float* wide_W   = (const float*)d_in[5];
    const float* wide_b   = (const float*)d_in[6];
    const float* bn_gamma = (const float*)d_in[7];
    const float* bn_beta  = (const float*)d_in[8];
    const float* router_W = (const float*)d_in[9];
    const float* exp_W1   = (const float*)d_in[10];
    const float* exp_b1   = (const float*)d_in[11];
    const float* exp_W2   = (const float*)d_in[12];
    const float* exp_b2   = (const float*)d_in[13];
    const float* d0_W     = (const float*)d_in[14];
    const float* d0_b     = (const float*)d_in[15];
    const float* d1_W     = (const float*)d_in[16];
    const float* d1_b     = (const float*)d_in[17];
    const float* out_W    = (const float*)d_in[18];
    const float* out_b    = (const float*)d_in[19];

    float* out = (float*)d_out;                 // [0..16383] = sigmoid out, [16384] = aux
    char*  ws  = (char*)d_ws;
    const size_t MiB = 1ull << 20;

    bf16*  ctx    = (bf16*)(ws + 0);            // 32 MiB   (later reused as moe)
    float* probs  = (float*)(ws + 32 * MiB);    // 1 MiB
    float* topv   = (float*)(ws + 33 * MiB);
    int*   topi   = (int*)  (ws + 33 * MiB + 128 * 1024);
    int*   slot   = (int*)  (ws + 33 * MiB + 256 * 1024);
    int*   counts = (int*)  (ws + 33 * MiB + 384 * 1024);
    bf16*  deepWt = (bf16*)(ws + 34 * MiB);     // (512,1024)
    bf16*  wideWt = (bf16*)(ws + 35 * MiB);     // (512,64)
    bf16*  e1Wt   = (bf16*)(ws + 36 * MiB);     // 16 x (1024,1024), 32 MiB
    bf16*  e2Wt   = (bf16*)(ws + 68 * MiB);     // 32 MiB
    bf16*  d0Wt   = (bf16*)(ws + 100 * MiB);    // (3072,1024), 6 MiB
    bf16*  d1Wt   = (bf16*)(ws + 106 * MiB);    // (1024,3072), 6 MiB
    bf16*  buf    = (bf16*)(ws + 112 * MiB);    // 128 MiB (later y, then h2 f32)
    bf16*  h      = (bf16*)(ws + 240 * MiB);    // 128 MiB (later h1)
    bf16*  ybuf   = buf;                        // alias: y over buf
    bf16*  moe    = ctx;                        // alias: moe over ctx
    bf16*  h1     = h;                          // alias: h1 over h (96 MiB)
    float* h2     = (float*)buf;                // alias: h2 f32 over buf (64 MiB)

    dim3 tb(32, 8);

    // 1) Transpose + convert weights to bf16 [N,K]
    transpose_f32_to_bf16<<<dim3(DHALF/32, DD/32, 1),  tb, 0, stream>>>(deep_W, deepWt, DD, DHALF, 0, 0);
    transpose_f32_to_bf16<<<dim3(DHALF/32, NWID/32, 1),tb, 0, stream>>>(wide_W, wideWt, NWID, DHALF, 0, 0);
    transpose_f32_to_bf16<<<dim3(DD/32, DD/32, NEXP),  tb, 0, stream>>>(exp_W1, e1Wt, DD, DD,
                                                                        (long long)DD*DD, (long long)DD*DD);
    transpose_f32_to_bf16<<<dim3(DD/32, DD/32, NEXP),  tb, 0, stream>>>(exp_W2, e2Wt, DD, DD,
                                                                        (long long)DD*DD, (long long)DD*DD);
    transpose_f32_to_bf16<<<dim3(N3/32, DD/32, 1),     tb, 0, stream>>>(d0_W, d0Wt, DD, N3, 0, 0);
    transpose_f32_to_bf16<<<dim3(DD/32, N3/32, 1),     tb, 0, stream>>>(d1_W, d1Wt, N3, DD, 0, 0);

    // 2) Zero expert capacity buffer
    {
        long long n8 = ((long long)NEXP * CAP * DD) / 8;   // 8,388,608
        zero_bf16<<<(unsigned)(n8 / 256), 256, 0, stream>>>(buf, n8);
    }

    // 3) Context head
    embed_gemm_wmma<<<dim3(DHALF/BNT, BQ/BM), 256, GEMM_SMEM + 4096, stream>>>(
        deep_in, embed, deepWt, deep_b, ctx);
    wide_kernel<<<BQ, 128, 0, stream>>>(wide_in, wideWt, wide_b, bn_gamma, bn_beta, ctx);

    // 4) Routing
    router_kernel<<<BQ, 128, 0, stream>>>(ctx, router_W, probs, topv, topi);
    scan_kernel<<<1, 1024, 0, stream>>>(topi, slot, counts);
    scatter_kernel<<<2 * BQ, 256, 0, stream>>>(ctx, slot, buf);

    // 5) Expert FFN (batched over 16 experts)
    gemm_bf16_wmma<<<dim3(DD/BNT, CAP/BM, NEXP), 256, GEMM_SMEM, stream>>>(
        buf, e1Wt, exp_b1, h, nullptr,
        CAP, DD, DD, DD,
        (long long)CAP*DD, (long long)DD*DD, DD, (long long)CAP*DD, 1);
    gemm_bf16_wmma<<<dim3(DD/BNT, CAP/BM, NEXP), 256, GEMM_SMEM, stream>>>(
        h, e2Wt, exp_b2, ybuf, nullptr,
        CAP, DD, DD, DD,
        (long long)CAP*DD, (long long)DD*DD, DD, (long long)CAP*DD, 0);

    // 6) Combine (weighted top-2 gather)  -> moe (aliases ctx)
    combine_kernel<<<BQ, 256, 0, stream>>>(ybuf, slot, topv, moe);

    // 7) Dense head
    gemm_bf16_wmma<<<dim3(N3/BNT, BQ/BM, 1), 256, GEMM_SMEM, stream>>>(
        moe, d0Wt, d0_b, h1, nullptr,
        BQ, N3, DD, N3, 0, 0, 0, 0, 1);
    gemm_bf16_wmma<<<dim3(DD/BNT, BQ/BM, 1), 256, GEMM_SMEM, stream>>>(
        h1, d1Wt, d1_b, nullptr, h2,
        BQ, DD, N3, DD, 0, 0, 0, 0, 1);

    // 8) Output + aux
    final_kernel<<<BQ, 256, 0, stream>>>(h2, out_W, out_b, out);
    aux_kernel<<<1, 256, 0, stream>>>(probs, counts, out + BQ);
}
